// FactorPredictor_40931038331019
// MI455X (gfx1250) — matrix operand order repaired and checked by reading, not compiled, not run on hardware
//
#include <hip/hip_runtime.h>
#include <hip/hip_bf16.h>

#define BS   65536
#define HD   32
#define F    64
#define H    32
#define FLAT 2048

#define NWG_A       128
#define WAVES_A     4
#define THREADS_A   (WAVES_A * 32)
#define CHUNKS      (BS / 32)            // 2048 chunks of 32 rows
#define TOTAL_WAVES (NWG_A * WAVES_A)    // 512 waves -> 4 chunks each
#define NPART       TOTAL_WAVES          // per-wave P1/P2 partials

// padded LDS strides (bank-conflict avoidance)
#define KF_STR 36   // f32, 144B rows: 16B-aligned float4 reads, odd-ish dword stride
#define BF_STR 34   // bf16, 17-dword stride: conflict-free lane*stride reads
#define W_STR  66   // bf16, 33-dword stride for the w staging

typedef __bf16 v16bf __attribute__((ext_vector_type(16)));
typedef float  v8f   __attribute__((ext_vector_type(8)));

// ---------------------------------------------------------------------------
__device__ __forceinline__ v8f wmma_bf16(v16bf a, v16bf b, v8f c) {
  return __builtin_amdgcn_wmma_f32_16x16x32_bf16(false, a, false, b,
                                                 (short)0, c, false, false);
}

// A-fragment (16x32 MxK) from row-major bf16 LDS with given stride.
// lane L -> row m = L&15, K-base kb = (L>>4)*8;
// elems 0..7 -> K=kb..kb+7, elems 8..15 -> K=16+kb..16+kb+7.
__device__ __forceinline__ v16bf loadA_lds(const __bf16* p, int stride, int lane) {
  const int m = lane & 15, kb = (lane >> 4) * 8;
  v16bf a;
#pragma unroll
  for (int j = 0; j < 8; ++j) a[j] = p[m * stride + kb + j];
#pragma unroll
  for (int j = 0; j < 8; ++j) a[8 + j] = p[m * stride + 16 + kb + j];
  return a;
}

// A-fragment of transposed matrix: A[m][k] = p[k*W_STR + col0 + m]  (A = w^T).
__device__ __forceinline__ v16bf loadA_trans(const __bf16* p, int col0, int lane) {
  const int m = lane & 15, kb = (lane >> 4) * 8;
  v16bf a;
#pragma unroll
  for (int j = 0; j < 8; ++j) a[j] = p[(kb + j) * W_STR + col0 + m];
#pragma unroll
  for (int j = 0; j < 8; ++j) a[8 + j] = p[(16 + kb + j) * W_STR + col0 + m];
  return a;
}

// A-fragment of a 16-row x tile straight from global fp32 (bf16 convert).
__device__ __forceinline__ v16bf loadA_x(const float* __restrict__ x,
                                         int row0, int lane) {
  const int m = lane & 15, kb = (lane >> 4) * 8;
  const float* px = x + (size_t)(row0 + m) * HD;
  v16bf a;
#pragma unroll
  for (int j = 0; j < 8; ++j) a[j] = (__bf16)px[kb + j];
#pragma unroll
  for (int j = 0; j < 8; ++j) a[8 + j] = (__bf16)px[16 + kb + j];
  return a;
}

// B-fragment (32x16 KxN): lane L -> N = n0+(L&15), elems j -> K = 8*(L>>4)+j (j<8), 16+8*(L>>4)+j-8 (j>=8).
__device__ __forceinline__ v16bf loadB_lds(const __bf16* p, int stride, int n0,
                                           int lane) {
  v16bf b;
#pragma unroll
  for (int j = 0; j < 16; ++j) b[j] = p[(((j & 8) << 1) + ((lane >> 4) << 3) + (j & 7)) * stride + n0 + (lane & 15)];
  return b;
}

// ---------------------------------------------------------------------------
// Phase 1: per 32-row chunk per wave:
//   WMMA k/v projection -> stage k(f32)/v(bf16) -> lane-per-row normalize
//   (no shuffles) -> WMMA D = r@qn -> w = rcp(D+32e-6) -> WMMA P1/P2 accum.
// ---------------------------------------------------------------------------
__global__ void __launch_bounds__(THREADS_A) fp_phase1(
    const float* __restrict__ x, const float* __restrict__ kW,
    const float* __restrict__ kb, const float* __restrict__ vW,
    const float* __restrict__ vb, const float* __restrict__ query,
    float* __restrict__ ws1, float* __restrict__ ws2) {
  __shared__ float s_qninv[F];
  __shared__ __align__(16) float  s_kf[WAVES_A][32 * KF_STR];  // raw k rows
  __shared__ __align__(16) __bf16 s_r [WAVES_A][32 * BF_STR];  // relu(kn)
  __shared__ __align__(16) __bf16 s_rv[WAVES_A][32 * BF_STR];  // r*v
  __shared__ __align__(16) __bf16 s_v [WAVES_A][32 * BF_STR];  // v
  __shared__ __align__(16) __bf16 s_w [WAVES_A][32 * W_STR];   // 1/D

  const int tid  = threadIdx.x;
  const int lane = tid & 31;
  const int wid  = tid >> 5;

  if (tid < F) {
    float ss = 0.f;
    for (int h = 0; h < H; ++h) { float q = query[h * F + tid]; ss += q * q; }
    s_qninv[tid] = __builtin_amdgcn_rcpf(sqrtf(ss) + 1e-6f);
  }
  __syncthreads();

  // Loop-invariant B fragments, gathered from global (once per wave).
  v16bf b_k0, b_k1, b_v0, b_v1, b_q[4];
#pragma unroll
  for (int j = 0; j < 16; ++j) {
    b_k0[j] = (__bf16)kW[(lane & 15) * HD + ((j & 8) << 1) + ((lane >> 4) << 3) + (j & 7)];          // kW^T[k(j)][lane&15]
    b_k1[j] = (__bf16)kW[(16 + (lane & 15)) * HD + ((j & 8) << 1) + ((lane >> 4) << 3) + (j & 7)];
    b_v0[j] = (__bf16)vW[(lane & 15) * HD + ((j & 8) << 1) + ((lane >> 4) << 3) + (j & 7)];
    b_v1[j] = (__bf16)vW[(16 + (lane & 15)) * HD + ((j & 8) << 1) + ((lane >> 4) << 3) + (j & 7)];
  }
#pragma unroll
  for (int t = 0; t < 4; ++t)
#pragma unroll
    for (int j = 0; j < 16; ++j)
      b_q[t][j] = (__bf16)(query[(((j & 8) << 1) + ((lane >> 4) << 3) + (j & 7)) * F + 16 * t + (lane & 15)] * s_qninv[16 * t + (lane & 15)]);

  const int ncol  = lane & 15;
  const int mbase = (lane >> 4) * 8;
  const float kb0 = kb[ncol], kb1 = kb[ncol + 16];
  const float vb0 = vb[ncol], vb1 = vb[ncol + 16];

  const v8f zc = (v8f){0.f, 0.f, 0.f, 0.f, 0.f, 0.f, 0.f, 0.f};
  v8f p1[4][2], p2[4][2];
#pragma unroll
  for (int t = 0; t < 4; ++t)
#pragma unroll
    for (int ht = 0; ht < 2; ++ht) { p1[t][ht] = zc; p2[t][ht] = zc; }

  const int waveId = blockIdx.x * WAVES_A + wid;

#pragma unroll 1
  for (int c = waveId; c < CHUNKS; c += TOTAL_WAVES) {
    const int row0 = c * 32;
    if (c + TOTAL_WAVES < CHUNKS)  // prefetch next chunk's x rows
      __builtin_prefetch(x + (size_t)(c + TOTAL_WAVES) * 32 * HD + lane * HD, 0, 1);

    // --- GEMM1: k/v projection, stage k (f32) + v (bf16) -------------------
#pragma unroll
    for (int sub = 0; sub < 2; ++sub) {
      v16bf a_x = loadA_x(x, row0 + sub * 16, lane);
      v8f ck0 = wmma_bf16(a_x, b_k0, zc);
      v8f ck1 = wmma_bf16(a_x, b_k1, zc);
      v8f cv0 = wmma_bf16(a_x, b_v0, zc);
      v8f cv1 = wmma_bf16(a_x, b_v1, zc);
#pragma unroll
      for (int j = 0; j < 8; ++j) {
        const int rr = sub * 16 + mbase + j;
        s_kf[wid][rr * KF_STR + ncol]      = ck0[j] + kb0;
        s_kf[wid][rr * KF_STR + 16 + ncol] = ck1[j] + kb1;
        s_v [wid][rr * BF_STR + ncol]      = (__bf16)(cv0[j] + vb0);
        s_v [wid][rr * BF_STR + 16 + ncol] = (__bf16)(cv1[j] + vb1);
      }
    }

    // --- lane-per-row normalize: zero shuffles, zero barriers --------------
    {
      float kr[32];
      const float4* k4 = (const float4*)&s_kf[wid][lane * KF_STR];
#pragma unroll
      for (int q = 0; q < 8; ++q) {
        float4 t4 = k4[q];
        kr[4 * q] = t4.x; kr[4 * q + 1] = t4.y;
        kr[4 * q + 2] = t4.z; kr[4 * q + 3] = t4.w;
      }
      float ss = 0.f;
#pragma unroll
      for (int h = 0; h < H; ++h) ss = fmaf(kr[h], kr[h], ss);
      const float inv = __builtin_amdgcn_rcpf(__builtin_amdgcn_sqrtf(ss) + 1e-6f);
#pragma unroll
      for (int h = 0; h < H; ++h) {
        float r  = fmaxf(kr[h] * inv, 0.f);
        float vv = (float)s_v[wid][lane * BF_STR + h];
        s_r [wid][lane * BF_STR + h] = (__bf16)r;
        s_rv[wid][lane * BF_STR + h] = (__bf16)(r * vv);
      }
    }

    // --- GEMM2: D = r @ qn ; w = rcp(D + H*1e-6) ---------------------------
#pragma unroll
    for (int sub = 0; sub < 2; ++sub) {
      v16bf a_r = loadA_lds(&s_r[wid][sub * 16 * BF_STR], BF_STR, lane);
#pragma unroll
      for (int t = 0; t < 4; ++t) {
        v8f d = wmma_bf16(a_r, b_q[t], zc);
#pragma unroll
        for (int j = 0; j < 8; ++j) {
          const int rr = sub * 16 + mbase + j;
          s_w[wid][rr * W_STR + 16 * t + ncol] =
              (__bf16)__builtin_amdgcn_rcpf(d[j] + 3.2e-5f);
        }
      }
    }

    // --- GEMM3: P1 += w^T @ (r*v) ; P2 += w^T @ v  (K = 32 chunk rows) -----
#pragma unroll
    for (int ht = 0; ht < 2; ++ht) {
      v16bf b_rv = loadB_lds(&s_rv[wid][0], BF_STR, 16 * ht, lane);
      v16bf b_vv = loadB_lds(&s_v [wid][0], BF_STR, 16 * ht, lane);
#pragma unroll
      for (int t = 0; t < 4; ++t) {
        v16bf a_w = loadA_trans(&s_w[wid][0], 16 * t, lane);
        p1[t][ht] = wmma_bf16(a_w, b_rv, p1[t][ht]);
        p2[t][ht] = wmma_bf16(a_w, b_vv, p2[t][ht]);
      }
    }
  }

  // --- per-wave partial writeout (deterministic, no atomics) ---------------
#pragma unroll
  for (int t = 0; t < 4; ++t)
#pragma unroll
    for (int ht = 0; ht < 2; ++ht)
#pragma unroll
      for (int j = 0; j < 8; ++j) {
        const int idx = (16 * t + mbase + j) * H + 16 * ht + ncol;
        ws1[(size_t)waveId * FLAT + idx] = p1[t][ht][j];
        ws2[(size_t)waveId * FLAT + idx] = p2[t][ht][j];
      }
}

// ---------------------------------------------------------------------------
// Phase 2 (single WG): reduce 512 partials, f = qn*P1 + 1e-6*P2, LayerNorm,
// two MLP heads (2048->64 relu, 64->64, softplus for sigma).
// ---------------------------------------------------------------------------
__global__ void __launch_bounds__(256) fp_phase2(
    const float* __restrict__ query, const float* __restrict__ ln_g,
    const float* __restrict__ ln_b,
    const float* __restrict__ muW1, const float* __restrict__ mub1,
    const float* __restrict__ muW2, const float* __restrict__ mub2,
    const float* __restrict__ sgW1, const float* __restrict__ sgb1,
    const float* __restrict__ sgW2, const float* __restrict__ sgb2,
    const float* __restrict__ ws1, const float* __restrict__ ws2,
    float* __restrict__ out) {
  __shared__ float s_f[FLAT];
  __shared__ float s_fn[FLAT];
  __shared__ float s_qninv[F];
  __shared__ float s_red[256], s_red2[256];
  __shared__ float s_h1[F], s_h2[F];
  __shared__ float s_stat[2];
  const int tid = threadIdx.x;

  if (tid < F) {
    float ss = 0.f;
    for (int h = 0; h < H; ++h) { float q = query[h * F + tid]; ss += q * q; }
    s_qninv[tid] = 1.f / (sqrtf(ss) + 1e-6f);
  }
  __syncthreads();

  for (int i = tid; i < FLAT; i += 256) {
    float a1 = 0.f, a2 = 0.f;
    for (int g = 0; g < NPART; ++g) {
      a1 += ws1[(size_t)g * FLAT + i];
      a2 += ws2[(size_t)g * FLAT + i];
    }
    int f = i >> 5, h = i & 31;
    s_f[i] = query[h * F + f] * s_qninv[f] * a1 + 1e-6f * a2;
  }
  __syncthreads();

  // mean
  float part = 0.f;
  for (int i = tid; i < FLAT; i += 256) part += s_f[i];
  s_red[tid] = part;
  __syncthreads();
  for (int s = 128; s > 0; s >>= 1) {
    if (tid < s) s_red[tid] += s_red[tid + s];
    __syncthreads();
  }
  if (tid == 0) s_stat[0] = s_red[0] * (1.f / FLAT);
  __syncthreads();
  const float mean = s_stat[0];

  // biased variance
  part = 0.f;
  for (int i = tid; i < FLAT; i += 256) { float d = s_f[i] - mean; part += d * d; }
  s_red[tid] = part;
  __syncthreads();
  for (int s = 128; s > 0; s >>= 1) {
    if (tid < s) s_red[tid] += s_red[tid + s];
    __syncthreads();
  }
  if (tid == 0) s_stat[1] = 1.f / sqrtf(s_red[0] * (1.f / FLAT) + 1e-5f);
  __syncthreads();
  const float rstd = s_stat[1];

  for (int i = tid; i < FLAT; i += 256)
    s_fn[i] = (s_f[i] - mean) * rstd * ln_g[i] + ln_b[i];
  __syncthreads();

  // Layer 1 of both heads: 256 threads = 64 outputs x 4 K-slices of 512
  {
    const int j = tid & 63, p4 = tid >> 6;
    const float* m1 = muW1 + (size_t)j * FLAT + p4 * 512;
    const float* g1 = sgW1 + (size_t)j * FLAT + p4 * 512;
    const float* fn = s_fn + p4 * 512;
    float am = 0.f, as = 0.f;
    for (int i = 0; i < 512; ++i) {
      float fv = fn[i];
      am += fv * m1[i];
      as += fv * g1[i];
    }
    s_red[tid] = am;
    s_red2[tid] = as;
  }
  __syncthreads();
  if (tid < F) {
    float am = s_red[tid] + s_red[tid + 64] + s_red[tid + 128] + s_red[tid + 192] + mub1[tid];
    float as = s_red2[tid] + s_red2[tid + 64] + s_red2[tid + 128] + s_red2[tid + 192] + sgb1[tid];
    s_h1[tid] = fmaxf(am, 0.f);
    s_h2[tid] = fmaxf(as, 0.f);
  }
  __syncthreads();
  if (tid < F) {
    float mu = mub2[tid], sg = sgb2[tid];
    for (int j = 0; j < F; ++j) {
      mu += s_h1[j] * muW2[tid * F + j];
      sg += s_h2[j] * sgW2[tid * F + j];
    }
    float sp = (sg > 20.f) ? sg : ((sg < -20.f) ? expf(sg) : log1pf(expf(sg)));
    out[tid]     = mu;
    out[F + tid] = sp;
  }
}

// ---------------------------------------------------------------------------
extern "C" void kernel_launch(void* const* d_in, const int* in_sizes, int n_in,
                              void* d_out, int out_size, void* d_ws, size_t ws_size,
                              hipStream_t stream) {
  (void)in_sizes; (void)n_in; (void)out_size; (void)ws_size;
  const float* x     = (const float*)d_in[0];
  const float* kW    = (const float*)d_in[1];
  const float* kb    = (const float*)d_in[2];
  const float* vW    = (const float*)d_in[3];
  const float* vb    = (const float*)d_in[4];
  const float* query = (const float*)d_in[5];
  const float* ln_g  = (const float*)d_in[6];
  const float* ln_b  = (const float*)d_in[7];
  const float* muW1  = (const float*)d_in[8];
  const float* mub1  = (const float*)d_in[9];
  const float* muW2  = (const float*)d_in[10];
  const float* mub2  = (const float*)d_in[11];
  const float* sgW1  = (const float*)d_in[12];
  const float* sgb1  = (const float*)d_in[13];
  const float* sgW2  = (const float*)d_in[14];
  const float* sgb2  = (const float*)d_in[15];

  float* ws1 = (float*)d_ws;                        // [NPART][2048] P1 partials
  float* ws2 = ws1 + (size_t)NPART * FLAT;          // [NPART][2048] P2 partials

  fp_phase1<<<NWG_A, THREADS_A, 0, stream>>>(x, kW, kb, vW, vb, query, ws1, ws2);
  fp_phase2<<<1, 256, 0, stream>>>(query, ln_g, ln_b, muW1, mub1, muW2, mub2,
                                   sgW1, sgb1, sgW2, sgb2, ws1, ws2,
                                   (float*)d_out);
}
